// MiniMaxSparseMoE_27101243638158
// MI455X (gfx1250) — matrix operand order repaired and checked by reading, not compiled, not run on hardware
//
#include <hip/hip_runtime.h>
#include <hip/hip_bf16.h>

typedef __attribute__((ext_vector_type(16))) __bf16 v16bf;
typedef __attribute__((ext_vector_type(4)))  __bf16 v4bf;
typedef __attribute__((ext_vector_type(8)))  float  v8f;
typedef __attribute__((ext_vector_type(4)))  unsigned int u32x4;
typedef __attribute__((ext_vector_type(8)))  int i32x8;
typedef __attribute__((ext_vector_type(4)))  int i32x4;

#define T_TOK 2048
#define D_DIM 2048
#define I_DIM 5632
#define E_NUM 8

#define BM 128
#define BN 64
#define BK 32

// workspace layout (bytes)
#define WS_CNT 0
#define WS_OFF 256
#define WS_TID 512
#define WS_TW  (512 + E_NUM * T_TOK * 4)
#define WS_H   (512 + 2 * E_NUM * T_TOK * 4)
#define H_ROWS (T_TOK * 2 + 128)   // 4096 used rows + 128 pad rows

// ---------------------------------------------------------------------------
// Router: one wave32 per token. logits = x @ gate_w^T, top-2, softmax,
// scatter token into per-expert lists.
// ---------------------------------------------------------------------------
__global__ __launch_bounds__(256)
void moe_router_kernel(const float* __restrict__ x,
                       const float* __restrict__ gate_w,
                       int* __restrict__ cnt,
                       int* __restrict__ tok_id,
                       float* __restrict__ tok_w) {
  const int wid  = threadIdx.x >> 5;
  const int lane = threadIdx.x & 31;
  const int t    = blockIdx.x * 8 + wid;
  if (t >= T_TOK) return;

  float acc[E_NUM];
#pragma unroll
  for (int e = 0; e < E_NUM; ++e) acc[e] = 0.f;

  const float4* xr = (const float4*)(x + (size_t)t * D_DIM);
#pragma unroll 4
  for (int it = 0; it < D_DIM / 4 / 32; ++it) {
    float4 xv = xr[lane + it * 32];
#pragma unroll
    for (int e = 0; e < E_NUM; ++e) {
      float4 gv = ((const float4*)(gate_w + (size_t)e * D_DIM))[lane + it * 32];
      acc[e] += xv.x * gv.x + xv.y * gv.y + xv.z * gv.z + xv.w * gv.w;
    }
  }
#pragma unroll
  for (int e = 0; e < E_NUM; ++e) {
#pragma unroll
    for (int m = 16; m >= 1; m >>= 1) acc[e] += __shfl_xor(acc[e], m, 32);
  }

  if (lane == 0) {
    int e0 = 0; float v0 = acc[0];
#pragma unroll
    for (int e = 1; e < E_NUM; ++e) if (acc[e] > v0) { v0 = acc[e]; e0 = e; }
    int e1 = (e0 == 0) ? 1 : 0; float v1 = acc[e1];
#pragma unroll
    for (int e = 0; e < E_NUM; ++e)
      if (e != e0 && acc[e] > v1) { v1 = acc[e]; e1 = e; }
    float p  = __expf(v1 - v0);
    float w0 = 1.f / (1.f + p);
    float w1 = p / (1.f + p);
    int s0 = atomicAdd(&cnt[e0], 1);
    tok_id[e0 * T_TOK + s0] = t; tok_w[e0 * T_TOK + s0] = w0;
    int s1 = atomicAdd(&cnt[e1], 1);
    tok_id[e1 * T_TOK + s1] = t; tok_w[e1 * T_TOK + s1] = w1;
  }
}

// ---------------------------------------------------------------------------
// Exclusive prefix over E=8 expert counts -> row offsets in compacted h buffer
// ---------------------------------------------------------------------------
__global__ void moe_prefix_kernel(const int* __restrict__ cnt,
                                  int* __restrict__ off) {
  if (threadIdx.x == 0) {
    int s = 0;
    for (int e = 0; e < E_NUM; ++e) { off[e] = s; s += cnt[e]; }
  }
}

// ---------------------------------------------------------------------------
// Per-wave fragment loads matching CDNA5 WMMA VGPR layouts
// ---------------------------------------------------------------------------
__device__ __forceinline__ v16bf load_a_frag(const __bf16* base, int lm16,
                                             int lh) {
  // A 16x32 bf16: lane m = lm16; chunk0 K = lh*8..+7 ; chunk1 K = 16+lh*8..+7
  const __bf16* p0 = base + lm16 * BK + lh * 8;
  v16bf a;
#pragma unroll
  for (int q = 0; q < 8; ++q) { a[q] = p0[q]; a[8 + q] = p0[16 + q]; }
  return a;
}

__device__ __forceinline__ v16bf load_b_frag(const __bf16* sB, int lane,
                                             int ncol) {
  // B 32x16 bf16: lane n = lane%16; chunk0 K = lh*8..+7 ; chunk1 K = 16+lh*8..+7
  const __bf16* p = sB + ((lane >> 4) * 8) * BN + ncol + (lane & 15);
  v16bf b;
#pragma unroll
  for (int q = 0; q < 8; ++q) { b[q] = p[q * BN]; b[8 + q] = p[(16 + q) * BN]; }
  return b;
}

__device__ __forceinline__ float fast_silu_mul(float g, float u) {
  return g * __builtin_amdgcn_rcpf(1.f + __expf(-g)) * u;
}

__device__ __forceinline__ v4bf pack4(float4 v) {
  return (v4bf){(__bf16)v.x, (__bf16)v.y, (__bf16)v.z, (__bf16)v.w};
}

// ---------------------------------------------------------------------------
// GEMM1: h = silu(x @ Wg^T) * (x @ Wu^T) for this expert's token list.
// Register-prefetch pipeline: store stage k -> barrier -> global loads for
// stage k+1 overlap the 8 WMMAs of stage k.
// grid = (m_tiles, n_tiles, experts) so same-weight blocks are adjacent.
// ---------------------------------------------------------------------------
__global__ __launch_bounds__(256)
void moe_gemm1_kernel(const float* __restrict__ x,
                      const float* __restrict__ w_gate,
                      const float* __restrict__ w_up,
                      const int* __restrict__ cnt,
                      const int* __restrict__ off,
                      const int* __restrict__ tok_id,
                      __bf16* __restrict__ h_buf) {
  __shared__ __align__(16) __bf16 sA[BM * BK];
  __shared__ __align__(16) __bf16 sBg[BK * BN];
  __shared__ __align__(16) __bf16 sBu[BK * BN];
  __shared__ int sTok[BM];

  const int e     = blockIdx.z;
  const int cnt_e = cnt[e];
  const int m0    = blockIdx.x * BM;
  if (m0 >= cnt_e) return;
  const int roff = off[e];
  const int n0   = blockIdx.y * BN;

  if (threadIdx.x < BM) {
    const int lm = m0 + threadIdx.x;
    sTok[threadIdx.x] = (lm < cnt_e) ? tok_id[e * T_TOK + lm] : 0;
  }
  __syncthreads();

  const int wid  = threadIdx.x >> 5;
  const int lane = threadIdx.x & 31;
  const int wm   = (wid >> 1) * 32;
  const int wn   = (wid & 1) * 32;
  const int lm16 = lane & 15;
  const int lh   = lane >> 4;

  // per-thread staging geometry (uniform per thread across K loop)
  const float* aPtr[4]; int aIdx[4];
#pragma unroll
  for (int j = 0; j < 4; ++j) {
    const int lin = threadIdx.x + j * 256;
    const int m   = lin >> 3;
    const int kq  = (lin & 7) * 4;
    aPtr[j] = x + (size_t)sTok[m] * D_DIM + kq;
    aIdx[j] = m * BK + kq;
  }
  const float* bgPtr[2]; const float* buPtr[2]; int bN[2], bK[2];
#pragma unroll
  for (int j = 0; j < 2; ++j) {
    const int lin = threadIdx.x + j * 256;
    const int n   = lin >> 3;
    const int kq  = (lin & 7) * 4;
    const size_t wrow = ((size_t)e * I_DIM + n0 + n) * D_DIM + kq;
    bgPtr[j] = w_gate + wrow;
    buPtr[j] = w_up + wrow;
    bN[j] = n; bK[j] = kq;
  }

  v8f accg[2][2], accu[2][2];
#pragma unroll
  for (int i = 0; i < 2; ++i)
#pragma unroll
    for (int j = 0; j < 2; ++j) { accg[i][j] = (v8f){}; accu[i][j] = (v8f){}; }

  // prologue: prefetch stage 0
  float4 rA[4], rG[2], rU[2];
#pragma unroll
  for (int j = 0; j < 4; ++j) rA[j] = *(const float4*)(aPtr[j]);
#pragma unroll
  for (int j = 0; j < 2; ++j) {
    rG[j] = *(const float4*)(bgPtr[j]);
    rU[j] = *(const float4*)(buPtr[j]);
  }

  const int NK = D_DIM / BK;
  for (int kk = 0; kk < NK; ++kk) {
    __syncthreads();  // previous stage consumed
    // store stage kk (regs -> LDS with fp32->bf16 convert; A packed b64)
#pragma unroll
    for (int j = 0; j < 4; ++j)
      *(v4bf*)&sA[aIdx[j]] = pack4(rA[j]);
#pragma unroll
    for (int j = 0; j < 2; ++j) {
      sBg[(bK[j] + 0) * BN + bN[j]] = (__bf16)rG[j].x;
      sBg[(bK[j] + 1) * BN + bN[j]] = (__bf16)rG[j].y;
      sBg[(bK[j] + 2) * BN + bN[j]] = (__bf16)rG[j].z;
      sBg[(bK[j] + 3) * BN + bN[j]] = (__bf16)rG[j].w;
      sBu[(bK[j] + 0) * BN + bN[j]] = (__bf16)rU[j].x;
      sBu[(bK[j] + 1) * BN + bN[j]] = (__bf16)rU[j].y;
      sBu[(bK[j] + 2) * BN + bN[j]] = (__bf16)rU[j].z;
      sBu[(bK[j] + 3) * BN + bN[j]] = (__bf16)rU[j].w;
    }
    __syncthreads();
    // prefetch stage kk+1 while WMMAs run on stage kk
    if (kk + 1 < NK) {
      const int k0n = (kk + 1) * BK;
#pragma unroll
      for (int j = 0; j < 4; ++j) rA[j] = *(const float4*)(aPtr[j] + k0n);
#pragma unroll
      for (int j = 0; j < 2; ++j) {
        rG[j] = *(const float4*)(bgPtr[j] + k0n);
        rU[j] = *(const float4*)(buPtr[j] + k0n);
      }
    }

    v16bf a[2], bg[2], bu[2];
#pragma unroll
    for (int sm = 0; sm < 2; ++sm)
      a[sm] = load_a_frag(&sA[(wm + sm * 16) * BK], lm16, lh);
#pragma unroll
    for (int sn = 0; sn < 2; ++sn) {
      bg[sn] = load_b_frag(sBg, lane, wn + sn * 16);
      bu[sn] = load_b_frag(sBu, lane, wn + sn * 16);
    }
#pragma unroll
    for (int sm = 0; sm < 2; ++sm)
#pragma unroll
      for (int sn = 0; sn < 2; ++sn) {
        accg[sm][sn] = __builtin_amdgcn_wmma_f32_16x16x32_bf16(
            false, a[sm], false, bg[sn], (short)0, accg[sm][sn], false, false);
        accu[sm][sn] = __builtin_amdgcn_wmma_f32_16x16x32_bf16(
            false, a[sm], false, bu[sn], (short)0, accu[sm][sn], false, false);
      }
  }

  // epilogue: h = silu(g) * u -> bf16 h buffer (compacted rows)
#pragma unroll
  for (int sm = 0; sm < 2; ++sm)
#pragma unroll
    for (int i = 0; i < 8; ++i) {
      const int mloc = m0 + wm + sm * 16 + i + 8 * lh;
      if (mloc < cnt_e) {
        const size_t rb = (size_t)(roff + mloc) * I_DIM + n0 + wn + lm16;
#pragma unroll
        for (int sn = 0; sn < 2; ++sn) {
          const float h = fast_silu_mul(accg[sm][sn][i], accu[sm][sn][i]);
          h_buf[rb + sn * 16] = (__bf16)h;
        }
      }
    }
}

// ---------------------------------------------------------------------------
// GEMM2: y = h @ Wd^T, scaled by routing weight, atomic-accumulated to out.
// A operand (bf16 h tile, contiguous rows) is staged via the Tensor Data
// Mover: one tensor_load_to_lds per K-step, issued by wave 0, synchronized
// with s_wait_tensorcnt before the consuming barrier.
// ---------------------------------------------------------------------------
__global__ __launch_bounds__(256)
void moe_gemm2_kernel(const __bf16* __restrict__ h_buf,
                      const float* __restrict__ w_down,
                      const int* __restrict__ cnt,
                      const int* __restrict__ off,
                      const int* __restrict__ tok_id,
                      const float* __restrict__ tok_w,
                      float* __restrict__ out) {
  __shared__ __align__(16) __bf16 sA[BM * BK];
  __shared__ __align__(16) __bf16 sB[BK * BN];

  const int e     = blockIdx.z;
  const int cnt_e = cnt[e];
  const int m0    = blockIdx.x * BM;
  if (m0 >= cnt_e) return;
  const int roff = off[e];
  const int n0   = blockIdx.y * BN;

  const int wid  = threadIdx.x >> 5;
  const int lane = threadIdx.x & 31;
  const int wm   = (wid >> 1) * 32;
  const int wn   = (wid & 1) * 32;
  const int lm16 = lane & 15;
  const int lh   = lane >> 4;

  // B staging geometry + register prefetch
  const float* bPtr[2]; int bN[2], bK[2];
#pragma unroll
  for (int j = 0; j < 2; ++j) {
    const int lin = threadIdx.x + j * 256;
    const int n   = lin >> 3;
    const int kq  = (lin & 7) * 4;
    bPtr[j] = w_down + ((size_t)e * D_DIM + n0 + n) * I_DIM + kq;
    bN[j] = n; bK[j] = kq;
  }
  float4 rB[2];
#pragma unroll
  for (int j = 0; j < 2; ++j) rB[j] = *(const float4*)(bPtr[j]);

  // TDM descriptor (uniform): 2D tile, 128 rows x 32 elems of 2B, row stride I
  const unsigned lds_off = (unsigned)(unsigned long long)(uintptr_t)(&sA[0]);
  const unsigned long long gaBase =
      (unsigned long long)(uintptr_t)h_buf +
      2ull * ((size_t)(roff + m0) * I_DIM);
  const i32x8 g1 = {0x00010000,          // data_size = 1 (2 bytes)
                    (int)0xFFFF0000,     // tensor_dim0 lo16 = 0xFFFF (no OOB)
                    (int)0xFFFFFFFF,     // tensor_dim0 hi16, tensor_dim1 lo16
                    (int)0x0020FFFF,     // tensor_dim1 hi16, tile_dim0 = 32
                    (int)0x00000080,     // tile_dim1 = 128, tile_dim2 = 0
                    (int)I_DIM,          // tensor_dim0_stride lo32 = 5632
                    0, 0};
  const i32x4 gz4 = {0, 0, 0, 0};
  const i32x8 gz8 = {0, 0, 0, 0, 0, 0, 0, 0};

  v8f acc[2][2];
#pragma unroll
  for (int i = 0; i < 2; ++i)
#pragma unroll
    for (int j = 0; j < 2; ++j) acc[i][j] = (v8f){};

  const int NK = I_DIM / BK;
  for (int kk = 0; kk < NK; ++kk) {
    __syncthreads();  // previous stage consumed
    if (wid == 0) {
      const unsigned long long ga = gaBase + 2ull * (unsigned)(kk * BK);
      const u32x4 g0 = {1u,  // count = 1, user descriptor
                        lds_off, (unsigned)ga,
                        (unsigned)((ga >> 32) & 0x1FFFFFFu) | 0x80000000u};
      __builtin_amdgcn_tensor_load_to_lds(g0, g1, gz4, gz4, gz8, 0);
    }
    // store B stage kk (fp32 -> bf16 transposed [k][n])
#pragma unroll
    for (int j = 0; j < 2; ++j) {
      sB[(bK[j] + 0) * BN + bN[j]] = (__bf16)rB[j].x;
      sB[(bK[j] + 1) * BN + bN[j]] = (__bf16)rB[j].y;
      sB[(bK[j] + 2) * BN + bN[j]] = (__bf16)rB[j].z;
      sB[(bK[j] + 3) * BN + bN[j]] = (__bf16)rB[j].w;
    }
    // prefetch B stage kk+1
    if (kk + 1 < NK) {
      const int k0n = (kk + 1) * BK;
#pragma unroll
      for (int j = 0; j < 2; ++j) rB[j] = *(const float4*)(bPtr[j] + k0n);
    }
    if (wid == 0) __builtin_amdgcn_s_wait_tensorcnt(0);
    __syncthreads();

    v16bf a[2], b[2];
#pragma unroll
    for (int sm = 0; sm < 2; ++sm)
      a[sm] = load_a_frag(&sA[(wm + sm * 16) * BK], lm16, lh);
#pragma unroll
    for (int sn = 0; sn < 2; ++sn)
      b[sn] = load_b_frag(sB, lane, wn + sn * 16);
#pragma unroll
    for (int sm = 0; sm < 2; ++sm)
#pragma unroll
      for (int sn = 0; sn < 2; ++sn)
        acc[sm][sn] = __builtin_amdgcn_wmma_f32_16x16x32_bf16(
            false, a[sm], false, b[sn], (short)0, acc[sm][sn], false, false);
  }

  // epilogue: scale by routing weight, scatter-accumulate to out[token]
#pragma unroll
  for (int sm = 0; sm < 2; ++sm)
#pragma unroll
    for (int i = 0; i < 8; ++i) {
      const int mloc = m0 + wm + sm * 16 + i + 8 * lh;
      if (mloc < cnt_e) {
        const int   tok = tok_id[e * T_TOK + mloc];
        const float w   = tok_w[e * T_TOK + mloc];
        float* ob = &out[(size_t)tok * D_DIM + n0 + wn + lm16];
#pragma unroll
        for (int sn = 0; sn < 2; ++sn)
          atomicAdd(ob + sn * 16, acc[sm][sn][i] * w);
      }
    }
}

// ---------------------------------------------------------------------------
// Launch
// ---------------------------------------------------------------------------
extern "C" void kernel_launch(void* const* d_in, const int* in_sizes, int n_in,
                              void* d_out, int out_size, void* d_ws,
                              size_t ws_size, hipStream_t stream) {
  const float* x      = (const float*)d_in[0];
  const float* gate_w = (const float*)d_in[1];
  const float* w_gate = (const float*)d_in[2];
  const float* w_up   = (const float*)d_in[3];
  const float* w_down = (const float*)d_in[4];
  float* out = (float*)d_out;

  char* ws = (char*)d_ws;
  int*    cnt   = (int*)(ws + WS_CNT);
  int*    off   = (int*)(ws + WS_OFF);
  int*    tokid = (int*)(ws + WS_TID);
  float*  tokw  = (float*)(ws + WS_TW);
  __bf16* h_buf = (__bf16*)(ws + WS_H);

  (void)hipMemsetAsync(cnt, 0, E_NUM * sizeof(int), stream);
  (void)hipMemsetAsync(out, 0, (size_t)out_size * sizeof(float), stream);

  moe_router_kernel<<<T_TOK / 8, 256, 0, stream>>>(x, gate_w, cnt, tokid, tokw);
  moe_prefix_kernel<<<1, 32, 0, stream>>>(cnt, off);

  dim3 g1(T_TOK / BM, I_DIM / BN, E_NUM);  // 16 x 88 x 8 (m fastest)
  moe_gemm1_kernel<<<g1, 256, 0, stream>>>(x, w_gate, w_up, cnt, off, tokid,
                                           h_buf);
  dim3 g2(T_TOK / BM, D_DIM / BN, E_NUM);  // 16 x 32 x 8
  moe_gemm2_kernel<<<g2, 256, 0, stream>>>(h_buf, w_down, cnt, off, tokid,
                                           tokw, out);
}
